// CardiacMamba_13159779795496
// MI455X (gfx1250) — compile-verified, matrix operand order checked by reading, not run on hardware
//
#include <hip/hip_runtime.h>
#include <hip/hip_bf16.h>

// ---- problem constants (from reference) ----
#define HID     256
#define DI      512
#define NH      8
#define HD      64
#define DS      16
#define DCONV   4
#define CONV_CH 544         // DI + 2*DS
#define PROJ    1064        // 2*DI + 2*DS + NH
#define BB      64
#define TTLEN   512
#define MTOK    (BB*TTLEN)  // 32768 tokens
#define NPAD1   1088        // PROJ padded to multiple of 64
#define MBLK    32          // rows of C per block (two 16-row WMMA sub-tiles)

typedef __attribute__((ext_vector_type(16))) __bf16 v16bf;
typedef __attribute__((ext_vector_type(8)))  float  v8f;

union FragBF { v16bf v; unsigned int u[8]; };

__device__ __forceinline__ unsigned short f2bf(float f) {
  unsigned int u = __float_as_uint(f);
  unsigned int r = u + 0x7FFFu + ((u >> 16) & 1u);   // round-to-nearest-even
  return (unsigned short)(r >> 16);
}
__device__ __forceinline__ float siluf(float x) { return x / (1.0f + expf(-x)); }
__device__ __forceinline__ float softplusf(float x) {
  return (x > 0.0f) ? x + log1pf(expf(-x)) : log1pf(expf(x));
}

// ---- weight convert + transpose: W (K x N, f32 row-major) -> Wt (Npad x K, bf16) ----
__global__ void k_convert_transpose(const float* __restrict__ W,
                                    unsigned short* __restrict__ Wt,
                                    int K, int N, int Npad) {
  long total = (long)Npad * K;
  for (long i = (long)blockIdx.x * blockDim.x + threadIdx.x; i < total;
       i += (long)gridDim.x * blockDim.x) {
    int n = (int)(i / K), k = (int)(i % K);
    float v = (n < N) ? W[(size_t)k * N + n] : 0.0f;
    Wt[i] = f2bf(v);
  }
}

// ---- rmsnorm(tokens) -> bf16 activation buffer ----
__global__ void k_rmsnorm_bf16(const float* __restrict__ X, const float* __restrict__ w,
                               unsigned short* __restrict__ O) {
  __shared__ float red[HID];
  int m = blockIdx.x, i = threadIdx.x;
  float x = X[(size_t)m * HID + i];
  red[i] = x * x; __syncthreads();
  for (int s = HID / 2; s > 0; s >>= 1) { if (i < s) red[i] += red[i + s]; __syncthreads(); }
  float inv = rsqrtf(red[0] * (1.0f / HID) + 1e-5f);
  O[(size_t)m * HID + i] = f2bf(x * inv * w[i]);
}

// ---- WMMA bf16 GEMM: C[M,N] = A[M,K] * Bt[Npad,K]^T (+bias)(+resid) ----
// Block: 256 threads (8 waves). A panel of 32 contiguous rows staged in LDS once.
// Each wave computes a 32x64 tile: 2 A-fragments x 4 B-fragments -> 8 WMMAs / k-step.
// Requirements: M % 32 == 0, K % 32 == 0, K <= 512, Npad % 64 == 0, Npad rows of Bt valid.
__global__ void k_gemm_bf16_wmma(const unsigned short* __restrict__ A,
                                 const unsigned short* __restrict__ Bt,
                                 const float* __restrict__ bias,
                                 const float* __restrict__ resid,
                                 float* __restrict__ C,
                                 int M, int N, int Npad, int K) {
  __shared__ unsigned int sA[MBLK * 256];           // 32 rows x (K/2 dwords), K<=512 -> 32KB
  const int kd = K >> 1;                            // dwords per row
  const int m0 = blockIdx.y * MBLK;

  // cooperative stage of A rows m0..m0+31 (contiguous in memory) into LDS
  {
    const unsigned int* gA = (const unsigned int*)(A + (size_t)m0 * K);
    const int total = MBLK * kd;                    // 16*K dwords
    for (int idx = (int)threadIdx.x; idx < total; idx += 256) sA[idx] = gA[idx];
  }
  __syncthreads();

  const int lane  = (int)(threadIdx.x & 31u);
  const int wave  = (int)(threadIdx.x >> 5);
  const int chunk = blockIdx.x * 8 + wave;          // 64 columns per wave
  const int n0    = chunk * 64;
  if (n0 >= Npad) return;                           // wave-uniform exit (after barrier)

  const int row  = lane & 15;
  const int ksel = (lane & 16) ? 4 : 0;             // dword offset for K-half per lane group
  const int rb0  = row * kd;                        // LDS row base, sub-tile 0
  const int rb1  = (16 + row) * kd;                 // LDS row base, sub-tile 1

  const unsigned int* bp0 = (const unsigned int*)(Bt + (size_t)(n0      + row) * K);
  const unsigned int* bp1 = (const unsigned int*)(Bt + (size_t)(n0 + 16 + row) * K);
  const unsigned int* bp2 = (const unsigned int*)(Bt + (size_t)(n0 + 32 + row) * K);
  const unsigned int* bp3 = (const unsigned int*)(Bt + (size_t)(n0 + 48 + row) * K);

  v8f a00 = {}, a01 = {}, a02 = {}, a03 = {};       // m-sub-tile 0, n-tiles 0..3
  v8f a10 = {}, a11 = {}, a12 = {}, a13 = {};       // m-sub-tile 1, n-tiles 0..3

  for (int kk = 0; kk < K; kk += 32) {
    const int base = (kk >> 1) + ksel;
    FragBF fa0, fa1;
    #pragma unroll
    for (int i = 0; i < 4; ++i) {
      fa0.u[i]     = sA[rb0 + base + i];
      fa0.u[i + 4] = sA[rb0 + base + 8 + i];
      fa1.u[i]     = sA[rb1 + base + i];
      fa1.u[i + 4] = sA[rb1 + base + 8 + i];
    }
    FragBF b;
    #pragma unroll
    for (int i = 0; i < 4; ++i) { b.u[i] = bp0[base + i]; b.u[i + 4] = bp0[base + 8 + i]; }
    __builtin_prefetch((const void*)(bp0 + base + 16), 0, 1);
    a00 = __builtin_amdgcn_wmma_f32_16x16x32_bf16(false, fa0.v, false, b.v, (short)0, a00, false, false);
    a10 = __builtin_amdgcn_wmma_f32_16x16x32_bf16(false, fa1.v, false, b.v, (short)0, a10, false, false);
    #pragma unroll
    for (int i = 0; i < 4; ++i) { b.u[i] = bp1[base + i]; b.u[i + 4] = bp1[base + 8 + i]; }
    a01 = __builtin_amdgcn_wmma_f32_16x16x32_bf16(false, fa0.v, false, b.v, (short)0, a01, false, false);
    a11 = __builtin_amdgcn_wmma_f32_16x16x32_bf16(false, fa1.v, false, b.v, (short)0, a11, false, false);
    #pragma unroll
    for (int i = 0; i < 4; ++i) { b.u[i] = bp2[base + i]; b.u[i + 4] = bp2[base + 8 + i]; }
    a02 = __builtin_amdgcn_wmma_f32_16x16x32_bf16(false, fa0.v, false, b.v, (short)0, a02, false, false);
    a12 = __builtin_amdgcn_wmma_f32_16x16x32_bf16(false, fa1.v, false, b.v, (short)0, a12, false, false);
    #pragma unroll
    for (int i = 0; i < 4; ++i) { b.u[i] = bp3[base + i]; b.u[i + 4] = bp3[base + 8 + i]; }
    a03 = __builtin_amdgcn_wmma_f32_16x16x32_bf16(false, fa0.v, false, b.v, (short)0, a03, false, false);
    a13 = __builtin_amdgcn_wmma_f32_16x16x32_bf16(false, fa1.v, false, b.v, (short)0, a13, false, false);
  }

  const int nl = lane & 15;
  const int mhalf = (lane & 16) ? 8 : 0;
  #define STORE_TILE(ACC, SI, TI) do {                               \
    int n_ = n0 + (TI) * 16 + nl;                                    \
    if (n_ < N) {                                                    \
      float bv_ = bias ? bias[n_] : 0.0f;                            \
      int mb_ = m0 + (SI) * 16 + mhalf;                              \
      _Pragma("unroll")                                              \
      for (int r_ = 0; r_ < 8; ++r_) {                               \
        size_t idx_ = (size_t)(mb_ + r_) * N + n_;                   \
        float v_ = ACC[r_] + bv_;                                    \
        if (resid) v_ += resid[idx_];                                \
        C[idx_] = v_;                                                \
      }                                                              \
    }                                                                \
  } while (0)
  STORE_TILE(a00, 0, 0); STORE_TILE(a01, 0, 1); STORE_TILE(a02, 0, 2); STORE_TILE(a03, 0, 3);
  STORE_TILE(a10, 1, 0); STORE_TILE(a11, 1, 1); STORE_TILE(a12, 1, 2); STORE_TILE(a13, 1, 3);
  #undef STORE_TILE
}

// ---- causal depthwise conv (width 4) + SiLU over xBC channels ----
__global__ void k_conv_silu(const float* __restrict__ Z, const float* __restrict__ cw,
                            const float* __restrict__ cb, float* __restrict__ XC) {
  long total = (long)MTOK * CONV_CH;
  for (long i = (long)blockIdx.x * blockDim.x + threadIdx.x; i < total;
       i += (long)gridDim.x * blockDim.x) {
    int  c = (int)(i % CONV_CH);
    long m = i / CONV_CH;
    int  t = (int)(m % TTLEN);
    float s = cb[c];
    #pragma unroll
    for (int j = 0; j < DCONV; ++j) {
      int tt = t - (DCONV - 1) + j;
      if (tt >= 0) s += cw[j * CONV_CH + c] * Z[(m - (DCONV - 1 - j)) * (long)PROJ + DI + c];
    }
    XC[i] = siluf(s);
  }
}

// ---- dt = softplus(dt_raw + dt_bias) ----
__global__ void k_dt(const float* __restrict__ Z, const float* __restrict__ dtb,
                     float* __restrict__ DT) {
  int i = blockIdx.x * blockDim.x + threadIdx.x;
  if (i >= MTOK * NH) return;
  int h = i & (NH - 1);
  long m = i >> 3;
  float x = Z[m * (long)PROJ + (DI + CONV_CH) + h] + dtb[h];
  DT[i] = softplusf(x);
}

// ---- SSM scan: one block per (b,h); 64 threads (one per head-dim); h[16] in regs ----
__global__ void k_ssm_scan(const float* __restrict__ XC, const float* __restrict__ DT,
                           const float* __restrict__ A_log, const float* __restrict__ Dp,
                           float* __restrict__ Y) {
  const int bh = blockIdx.x;
  const int b  = bh >> 3, h = bh & 7;
  const int d  = threadIdx.x;
  const float A  = -expf(A_log[h]);
  const float Dv = Dp[h];
  float hs[DS];
  #pragma unroll
  for (int s = 0; s < DS; ++s) hs[s] = 0.0f;
  __shared__ float sB[DS], sC[DS], sdt;
  for (int t = 0; t < TTLEN; ++t) {
    const size_t m = (size_t)b * TTLEN + t;
    if (d < DS)  { sB[d] = XC[m * CONV_CH + DI + d]; sC[d] = XC[m * CONV_CH + DI + DS + d]; }
    if (d == 0)  sdt = DT[m * NH + h];
    __syncthreads();
    const float dt  = sdt;
    const float dA  = expf(dt * A);
    const float x   = XC[m * CONV_CH + h * HD + d];
    const float dtx = dt * x;
    float y = 0.0f;
    #pragma unroll
    for (int s = 0; s < DS; ++s) { hs[s] = hs[s] * dA + dtx * sB[s]; y += hs[s] * sC[s]; }
    Y[m * DI + h * HD + d] = y + Dv * x;
    __syncthreads();
  }
}

// ---- y = rmsnorm(y * silu(z), gnorm_w) -> bf16 ----
__global__ void k_gate_norm(const float* __restrict__ Y, const float* __restrict__ Z,
                            const float* __restrict__ gw, unsigned short* __restrict__ O) {
  __shared__ float red[256];
  int m = blockIdx.x, i = threadIdx.x;
  float vals[2]; float acc = 0.0f;
  #pragma unroll
  for (int j = 0; j < 2; ++j) {
    int c = i + j * 256;
    float z = Z[(size_t)m * PROJ + c];
    float y = Y[(size_t)m * DI + c];
    float v = y * siluf(z);
    vals[j] = v; acc += v * v;
  }
  red[i] = acc; __syncthreads();
  for (int s = 128; s > 0; s >>= 1) { if (i < s) red[i] += red[i + s]; __syncthreads(); }
  float inv = rsqrtf(red[0] * (1.0f / DI) + 1e-5f);
  #pragma unroll
  for (int j = 0; j < 2; ++j) {
    int c = i + j * 256;
    O[(size_t)m * DI + c] = f2bf(vals[j] * inv * gw[c]);
  }
}

// ---- padded = shift(temporal_out) with init_state at t==0 -> bf16 ----
__global__ void k_shift_bf16(const float* __restrict__ TO, const float* __restrict__ init,
                             unsigned short* __restrict__ P) {
  long total = (long)MTOK * HID;
  for (long i = (long)blockIdx.x * blockDim.x + threadIdx.x; i < total;
       i += (long)gridDim.x * blockDim.x) {
    int  c = (int)(i % HID);
    long m = i / HID;
    int  t = (int)(m % TTLEN);
    float v = (t == 0) ? init[c] : TO[(m - 1) * (long)HID + c];
    P[i] = f2bf(v);
  }
}

// ---- per-token Kalman head -> v_post into d_out[0..MTOK) ----
__global__ void k_kalman(const float* __restrict__ H1, const float* __restrict__ H2,
                         const float* __restrict__ zt,
                         const float* __restrict__ kp_w2, const float* __restrict__ kp_b2,
                         const float* __restrict__ kg_w1, const float* __restrict__ kg_w2,
                         const float* __restrict__ kg_b2,
                         const float* __restrict__ km_w1, const float* __restrict__ km_b1,
                         const float* __restrict__ km_w2, const float* __restrict__ km_b2,
                         const float* __restrict__ kW, float* __restrict__ out) {
  int m = blockIdx.x * blockDim.x + threadIdx.x;
  if (m >= MTOK) return;
  float z = zt[m];
  float a1 = 0.0f;
  for (int j = 0; j < 64; ++j) a1 += siluf(H1[(size_t)m * 64 + j]) * kp_w2[j];
  float v_prior = softplusf(a1 + kp_b2[0]);
  const float* kgl = kg_w1 + (size_t)HID * 64;      // last row of (HID+1,64)
  float a2 = 0.0f;
  for (int j = 0; j < 64; ++j) a2 += siluf(H2[(size_t)m * 64 + j] + z * kgl[j]) * kg_w2[j];
  float Kt = 1.0f / (1.0f + expf(-(a2 + kg_b2[0])));
  float a3 = 0.0f;
  for (int j = 0; j < 32; ++j) a3 += siluf(z * km_w1[j] + km_b1[j]) * km_w2[j];
  float mt = softplusf(a3 + km_b2[0]);
  float vp = v_prior + Kt * (mt - kW[0] * v_prior);
  vp = fminf(fmaxf(vp, 1e-6f), 10.0f);
  out[m] = vp;
}

// ---- per-batch softmax pooling: pred_edv / pred_esv ----
__global__ void k_pool(const float* __restrict__ vpost, float* __restrict__ out) {
  __shared__ float red[256];
  int b = blockIdx.x, i = threadIdx.x;
  float v0 = vpost[(size_t)b * TTLEN + i];
  float v1 = vpost[(size_t)b * TTLEN + 256 + i];
  red[i] = v0 + v1; __syncthreads();
  for (int s = 128; s > 0; s >>= 1) { if (i < s) red[i] += red[i + s]; __syncthreads(); }
  float mean = red[0] * (1.0f / TTLEN); __syncthreads();
  float c0 = v0 - mean, c1 = v1 - mean;
  // softmax(+vc)
  red[i] = fmaxf(c0, c1); __syncthreads();
  for (int s = 128; s > 0; s >>= 1) { if (i < s) red[i] = fmaxf(red[i], red[i + s]); __syncthreads(); }
  float mx = red[0]; __syncthreads();
  float e0 = expf(c0 - mx), e1 = expf(c1 - mx);
  red[i] = e0 + e1; __syncthreads();
  for (int s = 128; s > 0; s >>= 1) { if (i < s) red[i] += red[i + s]; __syncthreads(); }
  float den = red[0]; __syncthreads();
  red[i] = e0 * v0 + e1 * v1; __syncthreads();
  for (int s = 128; s > 0; s >>= 1) { if (i < s) red[i] += red[i + s]; __syncthreads(); }
  float edv = red[0] / den; __syncthreads();
  // softmax(-vc): max(-c) = -min(c)
  red[i] = fminf(c0, c1); __syncthreads();
  for (int s = 128; s > 0; s >>= 1) { if (i < s) red[i] = fminf(red[i], red[i + s]); __syncthreads(); }
  float mn = red[0]; __syncthreads();
  float f0 = expf(mn - c0), f1 = expf(mn - c1);
  red[i] = f0 + f1; __syncthreads();
  for (int s = 128; s > 0; s >>= 1) { if (i < s) red[i] += red[i + s]; __syncthreads(); }
  float den2 = red[0]; __syncthreads();
  red[i] = f0 * v0 + f1 * v1; __syncthreads();
  for (int s = 128; s > 0; s >>= 1) { if (i < s) red[i] += red[i + s]; __syncthreads(); }
  if (i == 0) {
    out[MTOK + b]      = edv;
    out[MTOK + 64 + b] = red[0] / den2;
  }
}

extern "C" void kernel_launch(void* const* d_in, const int* in_sizes, int n_in,
                              void* d_out, int out_size, void* d_ws, size_t ws_size,
                              hipStream_t stream) {
  (void)in_sizes; (void)n_in; (void)out_size; (void)ws_size;
  const float* tokens  = (const float*)d_in[0];
  const float* z_t     = (const float*)d_in[1];
  const float* norm_w  = (const float*)d_in[2];
  const float* in_w    = (const float*)d_in[3];
  const float* in_b    = (const float*)d_in[4];
  const float* conv_w  = (const float*)d_in[5];
  const float* conv_b  = (const float*)d_in[6];
  const float* dt_bias = (const float*)d_in[7];
  const float* A_log   = (const float*)d_in[8];
  const float* Dp      = (const float*)d_in[9];
  const float* gnorm_w = (const float*)d_in[10];
  const float* out_w   = (const float*)d_in[11];
  const float* out_b   = (const float*)d_in[12];
  const float* kp_w1   = (const float*)d_in[13];
  const float* kp_b1   = (const float*)d_in[14];
  const float* kp_w2   = (const float*)d_in[15];
  const float* kp_b2   = (const float*)d_in[16];
  const float* km_w1   = (const float*)d_in[17];
  const float* km_b1   = (const float*)d_in[18];
  const float* km_w2   = (const float*)d_in[19];
  const float* km_b2   = (const float*)d_in[20];
  const float* kg_w1   = (const float*)d_in[21];
  const float* kg_b1   = (const float*)d_in[22];
  const float* kg_w2   = (const float*)d_in[23];
  const float* kg_b2   = (const float*)d_in[24];
  const float* kWs     = (const float*)d_in[25];
  const float* init_s  = (const float*)d_in[26];
  float* out = (float*)d_out;

  char* ws = (char*)d_ws;
  size_t off = 0;
  auto alloc = [&](size_t bytes) { size_t o = off; off += (bytes + 255) & ~(size_t)255; return o; };
  unsigned short* Wt1 = (unsigned short*)(ws + alloc((size_t)NPAD1 * HID * 2));
  unsigned short* Wt2 = (unsigned short*)(ws + alloc((size_t)HID * DI * 2));
  unsigned short* Wtp = (unsigned short*)(ws + alloc((size_t)64 * HID * 2));
  unsigned short* Wtg = (unsigned short*)(ws + alloc((size_t)64 * HID * 2));
  unsigned short* Abf = (unsigned short*)(ws + alloc((size_t)MTOK * HID * 2));
  float*          Zx  = (float*)         (ws + alloc((size_t)MTOK * PROJ * 4));
  float*          XC  = (float*)         (ws + alloc((size_t)MTOK * CONV_CH * 4));
  float*          DTb = (float*)         (ws + alloc((size_t)MTOK * NH * 4));
  float*          Yb  = (float*)         (ws + alloc((size_t)MTOK * DI * 4));
  unsigned short* Ybf = (unsigned short*)(ws + alloc((size_t)MTOK * DI * 2));
  float*          TO  = (float*)         (ws + alloc((size_t)MTOK * HID * 4));
  unsigned short* Pbf = (unsigned short*)(ws + alloc((size_t)MTOK * HID * 2));
  float*          H1  = (float*)         (ws + alloc((size_t)MTOK * 64 * 4));
  float*          H2  = (float*)         (ws + alloc((size_t)MTOK * 64 * 4));

  // weight prep (bf16, transposed to N x K)
  k_convert_transpose<<<512, 256, 0, stream>>>(in_w,  Wt1, HID, PROJ, NPAD1);
  k_convert_transpose<<<512, 256, 0, stream>>>(out_w, Wt2, DI,  HID,  HID);
  k_convert_transpose<<<64,  256, 0, stream>>>(kp_w1, Wtp, HID, 64,   64);
  k_convert_transpose<<<64,  256, 0, stream>>>(kg_w1, Wtg, HID, 64,   64);

  // mamba2 block
  k_rmsnorm_bf16<<<MTOK, HID, 0, stream>>>(tokens, norm_w, Abf);
  {
    dim3 g((NPAD1 / 64 + 7) / 8, MTOK / MBLK);
    k_gemm_bf16_wmma<<<g, 256, 0, stream>>>(Abf, Wt1, in_b, nullptr, Zx, MTOK, PROJ, NPAD1, HID);
  }
  k_conv_silu<<<(MTOK * CONV_CH + 255) / 256, 256, 0, stream>>>(Zx, conv_w, conv_b, XC);
  k_dt<<<(MTOK * NH + 255) / 256, 256, 0, stream>>>(Zx, dt_bias, DTb);
  k_ssm_scan<<<BB * NH, HD, 0, stream>>>(XC, DTb, A_log, Dp, Yb);
  k_gate_norm<<<MTOK, 256, 0, stream>>>(Yb, Zx, gnorm_w, Ybf);
  {
    dim3 g((HID / 64 + 7) / 8, MTOK / MBLK);
    k_gemm_bf16_wmma<<<g, 256, 0, stream>>>(Ybf, Wt2, out_b, tokens, TO, MTOK, HID, HID, DI);
  }

  // kalman head
  k_shift_bf16<<<(MTOK * HID + 255) / 256, 256, 0, stream>>>(TO, init_s, Pbf);
  {
    dim3 g(1, MTOK / MBLK);
    k_gemm_bf16_wmma<<<g, 256, 0, stream>>>(Pbf, Wtp, kp_b1, nullptr, H1, MTOK, 64, 64, HID);
    k_gemm_bf16_wmma<<<g, 256, 0, stream>>>(Pbf, Wtg, kg_b1, nullptr, H2, MTOK, 64, 64, HID);
  }
  k_kalman<<<(MTOK + 255) / 256, 256, 0, stream>>>(H1, H2, z_t, kp_w2, kp_b2, kg_w1, kg_w2,
                                                   kg_b2, km_w1, km_b1, km_w2, km_b2, kWs, out);
  k_pool<<<BB, 256, 0, stream>>>(out, out);
}
